// EncoderBlock_7859790152260
// MI455X (gfx1250) — compile-verified, matrix operand order checked by reading, not run on hardware
//
#include <hip/hip_runtime.h>
#include <hip/hip_fp16.h>

#define B_   8
#define N_   32768
#define C_   256
#define S_   64
#define NS_  8
#define R2_  16.0f
#define EPS_ 1e-5f

typedef __attribute__((ext_vector_type(16))) _Float16 v16h;
typedef __attribute__((ext_vector_type(2)))  _Float16 v2h;
typedef __attribute__((ext_vector_type(8)))  float    v8f;

// ---------------------------------------------------------------------------
// 1) Farthest point sampling: one workgroup per batch, dist array in LDS.
//    Sequential over K=64, parallel over N=32768 (1024 threads x 32 points).
// ---------------------------------------------------------------------------
__global__ void fps_kernel(const float* __restrict__ coor,
                           int* __restrict__ fps_idx,
                           float* __restrict__ sample_coor) {
    __shared__ float dist[N_];          // 128 KB (LDS, 320KB/WGP available)
    __shared__ float rv[1024];
    __shared__ int   ri[1024];
    __shared__ float curc[3];
    __shared__ int   cur;

    const int b   = blockIdx.x;
    const int tid = threadIdx.x;
    const float* cb = coor + (size_t)b * N_ * 3;

    for (int i = tid; i < N_; i += 1024) dist[i] = 1e10f;
    if (tid == 0) cur = 0;
    __syncthreads();

    for (int k = 0; k < S_; ++k) {
        if (tid == 0) {
            const int c = cur;
            fps_idx[b * S_ + k] = c;
            curc[0] = cb[c * 3 + 0];
            curc[1] = cb[c * 3 + 1];
            curc[2] = cb[c * 3 + 2];
            sample_coor[(b * S_ + k) * 3 + 0] = curc[0];
            sample_coor[(b * S_ + k) * 3 + 1] = curc[1];
            sample_coor[(b * S_ + k) * 3 + 2] = curc[2];
        }
        __syncthreads();
        const float cx = curc[0], cy = curc[1], cz = curc[2];

        float bv = -1.0f; int bi = 0x7fffffff;
        for (int i = tid; i < N_; i += 1024) {
            const float dx = cb[i * 3 + 0] - cx;
            const float dy = cb[i * 3 + 1] - cy;
            const float dz = cb[i * 3 + 2] - cz;
            const float d  = dx * dx + dy * dy + dz * dz;
            const float dm = fminf(dist[i], d);
            dist[i] = dm;
            if (dm > bv) { bv = dm; bi = i; }  // increasing i -> first-max kept
        }
        rv[tid] = bv; ri[tid] = bi;
        __syncthreads();
        for (int s = 512; s > 0; s >>= 1) {
            if (tid < s) {
                const float v2 = rv[tid + s]; const int i2 = ri[tid + s];
                if (v2 > rv[tid] || (v2 == rv[tid] && i2 < ri[tid])) {
                    rv[tid] = v2; ri[tid] = i2;
                }
            }
            __syncthreads();
        }
        if (tid == 0) cur = ri[0];
        __syncthreads();
    }
}

// ---------------------------------------------------------------------------
// 2) Convert Wq/Wk to f16 for WMMA.
// ---------------------------------------------------------------------------
__global__ void wconv_kernel(const float* __restrict__ Wq, const float* __restrict__ Wk,
                             _Float16* __restrict__ Wq16, _Float16* __restrict__ Wk16) {
    const int i = blockIdx.x * blockDim.x + threadIdx.x;
    if (i < C_ * C_) {
        Wq16[i] = (_Float16)Wq[i];
        Wk16[i] = (_Float16)Wk[i];
    }
}

// ---------------------------------------------------------------------------
// 3) Ball query: one wave32 per center, scan N in index order via ballot;
//    append first NS_ hits, pad with first hit (CUDA ball_query semantics).
// ---------------------------------------------------------------------------
__global__ void ballq_kernel(const float* __restrict__ coor,
                             const float* __restrict__ sample_coor,
                             int* __restrict__ nidx) {
    const int bs   = blockIdx.x;           // b*S + s
    const int b    = bs / S_;
    const int lane = threadIdx.x;
    const float* cb = coor + (size_t)b * N_ * 3;
    const float cx = sample_coor[bs * 3 + 0];
    const float cy = sample_coor[bs * 3 + 1];
    const float cz = sample_coor[bs * 3 + 2];

    int nb[NS_];
    int cnt = 0;
    for (int base = 0; base < N_ && cnt < NS_; base += 32) {
        const int i = base + lane;
        const float dx = cb[i * 3 + 0] - cx;
        const float dy = cb[i * 3 + 1] - cy;
        const float dz = cb[i * 3 + 2] - cz;
        const float d2 = dx * dx + dy * dy + dz * dz;
        unsigned mm = (unsigned)__ballot(d2 < R2_);   // wave32: low 32 bits
        while (mm && cnt < NS_) {                     // uniform serial append
            const int p = __builtin_ctz(mm);
            mm &= mm - 1;
            nb[cnt++] = base + p;
        }
    }
    const int pad = (cnt > 0) ? nb[0] : 0;
    for (int j = cnt; j < NS_; ++j) nb[j] = pad;
    if (lane < NS_) nidx[bs * NS_ + lane] = nb[lane];
}

// ---------------------------------------------------------------------------
// 4) Group: max-pool features over neighbors (= output1), diff_x, sample_x,
//    and mean relative coordinates diff_coor.
// ---------------------------------------------------------------------------
__global__ void group_kernel(const float* __restrict__ x, const float* __restrict__ coor,
                             const int* __restrict__ fps_idx,
                             const float* __restrict__ sample_coor,
                             const int* __restrict__ nidx,
                             float* __restrict__ out1, float* __restrict__ diff_x,
                             float* __restrict__ sample_x, float* __restrict__ diff_coor) {
    __shared__ int nb[NS_];
    __shared__ int fi;
    const int bs = blockIdx.x;
    const int b  = bs / S_;
    const int c  = threadIdx.x;
    if (c < NS_) nb[c] = nidx[bs * NS_ + c];
    if (c == 0)  fi = fps_idx[bs];
    __syncthreads();

    const float* xb = x + (size_t)b * N_ * C_;
    const float  sx = xb[(size_t)fi * C_ + c];
    float mx = -3.4e38f;
#pragma unroll
    for (int j = 0; j < NS_; ++j) mx = fmaxf(mx, xb[(size_t)nb[j] * C_ + c]);

    out1[(size_t)bs * C_ + c]     = mx;        // sample_x + diff_x == global max
    diff_x[(size_t)bs * C_ + c]   = mx - sx;
    sample_x[(size_t)bs * C_ + c] = sx;

    if (c < 3) {
        const float* cb = coor + (size_t)b * N_ * 3;
        const float  sc = sample_coor[bs * 3 + c];
        float acc = 0.0f;
#pragma unroll
        for (int j = 0; j < NS_; ++j) acc += cb[nb[j] * 3 + c] - sc;
        diff_coor[bs * 3 + c] = acc * (1.0f / NS_);
    }
}

// ---------------------------------------------------------------------------
// 5) LayerNorm (per row of 256) -> f16 activations for the WMMA GEMMs.
//    blockIdx < B*S : diff_x -> Aq ; blockIdx >= B*S : sample_x -> Ak.
// ---------------------------------------------------------------------------
__global__ void ln_kernel(const float* __restrict__ diff_x, const float* __restrict__ sample_x,
                          const float* __restrict__ gq, const float* __restrict__ bq,
                          const float* __restrict__ gk, const float* __restrict__ bk,
                          _Float16* __restrict__ Aq, _Float16* __restrict__ Ak) {
    __shared__ float red[C_];
    const int row   = blockIdx.x % (B_ * S_);
    const int which = blockIdx.x / (B_ * S_);
    const float* src = (which == 0 ? diff_x : sample_x) + (size_t)row * C_;
    const float* g   = (which == 0 ? gq : gk);
    const float* bb  = (which == 0 ? bq : bk);
    _Float16* dst    = (which == 0 ? Aq : Ak) + (size_t)row * C_;

    const int c = threadIdx.x;
    const float v = src[c];
    red[c] = v; __syncthreads();
    for (int s = 128; s > 0; s >>= 1) { if (c < s) red[c] += red[c + s]; __syncthreads(); }
    const float m = red[0] * (1.0f / C_);
    __syncthreads();
    const float d = v - m;
    red[c] = d * d; __syncthreads();
    for (int s = 128; s > 0; s >>= 1) { if (c < s) red[c] += red[c + s]; __syncthreads(); }
    const float var = red[0] * (1.0f / C_);
    dst[c] = (_Float16)(d * rsqrtf(var + EPS_) * g[c] + bb[c]);
}

// ---------------------------------------------------------------------------
// 6) WMMA GEMM: q = Aq @ Wq^T, k = Ak @ Wk^T.  1024 wave-jobs:
//    {q,k} x B x (M/16) x (N/16); K=256 -> 8 steps of 16x16x32 f16 WMMA.
//    Fragment gathers follow ISA 7.12.2 VGPR layouts (wave32).
// ---------------------------------------------------------------------------
__global__ void qk_gemm_kernel(const _Float16* __restrict__ Aq, const _Float16* __restrict__ Ak,
                               const _Float16* __restrict__ Wq16, const _Float16* __restrict__ Wk16,
                               _Float16* __restrict__ q16, _Float16* __restrict__ k16) {
    const int wave  = (blockIdx.x * blockDim.x + threadIdx.x) >> 5;
    const int lane  = threadIdx.x & 31;
    const int which = wave >> 9;          // 0 = q, 1 = k
    const int rem   = wave & 511;
    const int b     = rem >> 6;
    const int t     = rem & 63;
    const int mt    = t >> 4, nt = t & 15;

    const _Float16* A = (which ? Ak : Aq) + (size_t)b * S_ * C_;
    const _Float16* W = (which ? Wk16 : Wq16);
    _Float16*       D = (which ? k16 : q16) + (size_t)b * S_ * C_;

    const int H = lane >> 4, l = lane & 15;
    const int m = mt * 16 + l;            // A row for this lane
    const int n = nt * 16 + l;            // W row (= output column)

    v8f acc = {};
    for (int ks = 0; ks < C_ / 32; ++ks) {
        v16h a, w;
#pragma unroll
        for (int r = 0; r < 8; ++r) {
            const int k0 = ks * 32 + ((r & 4) ? 16 : 0) + H * 8 + (r & 3) * 2;
            const v2h av = *(const v2h*)(A + m * C_ + k0);
            const v2h wv = *(const v2h*)(W + n * C_ + k0);
            a[2 * r] = av[0]; a[2 * r + 1] = av[1];
            w[2 * r] = wv[0]; w[2 * r + 1] = wv[1];
        }
        acc = __builtin_amdgcn_wmma_f32_16x16x32_f16(false, a, false, w,
                                                     (short)0, acc, false, false);
    }
#pragma unroll
    for (int r = 0; r < 8; ++r)
        D[(mt * 16 + H * 8 + r) * C_ + nt * 16 + l] = (_Float16)acc[r];
}

// ---------------------------------------------------------------------------
// 7) Attention: scores = q @ k^T * 1/16 via WMMA (16 tiles/batch, one per
//    wave of a 512-thread block), softmax rows, coor2 = attn @ v with the
//    faithful transpose-view reinterpretation of diff_coor.
// ---------------------------------------------------------------------------
__global__ void attn_kernel(const _Float16* __restrict__ q16, const _Float16* __restrict__ k16,
                            const float* __restrict__ diff_coor,
                            const float* __restrict__ sample_coor,
                            float* __restrict__ out2) {
    __shared__ float sc[S_ * S_];
    __shared__ float vv[S_ * 3];
    const int b    = blockIdx.x;
    const int tid  = threadIdx.x;
    const int wave = tid >> 5, lane = tid & 31;
    const int mt   = wave >> 2, nt = wave & 3;

    const _Float16* Q = q16 + (size_t)b * S_ * C_;
    const _Float16* K = k16 + (size_t)b * S_ * C_;
    const int H = lane >> 4, l = lane & 15;
    const int m = mt * 16 + l;
    const int n = nt * 16 + l;

    v8f acc = {};
    for (int ks = 0; ks < C_ / 32; ++ks) {
        v16h a, w;
#pragma unroll
        for (int r = 0; r < 8; ++r) {
            const int k0 = ks * 32 + ((r & 4) ? 16 : 0) + H * 8 + (r & 3) * 2;
            const v2h av = *(const v2h*)(Q + m * C_ + k0);
            const v2h wv = *(const v2h*)(K + n * C_ + k0);
            a[2 * r] = av[0]; a[2 * r + 1] = av[1];
            w[2 * r] = wv[0]; w[2 * r + 1] = wv[1];
        }
        acc = __builtin_amdgcn_wmma_f32_16x16x32_f16(false, a, false, w,
                                                     (short)0, acc, false, false);
    }
#pragma unroll
    for (int r = 0; r < 8; ++r)
        sc[(mt * 16 + H * 8 + r) * S_ + nt * 16 + l] = acc[r] * 0.0625f;  // 1/sqrt(256)

    // v = transpose(diff_coor,(0,2,1)).reshape(B,S,3): flat e -> [e%64][e/64]
    if (tid < S_ * 3)
        vv[tid] = diff_coor[b * S_ * 3 + (tid % S_) * 3 + (tid / S_)];
    __syncthreads();

    if (tid < S_) {
        const int row = tid;
        float mx = -3.4e38f;
        for (int j = 0; j < S_; ++j) mx = fmaxf(mx, sc[row * S_ + j]);
        float sum = 0.0f;
        for (int j = 0; j < S_; ++j) {
            const float e = __expf(sc[row * S_ + j] - mx);
            sc[row * S_ + j] = e;      // row owned exclusively by this thread
            sum += e;
        }
        const float inv = 1.0f / sum;
#pragma unroll
        for (int d = 0; d < 3; ++d) {
            float a2 = 0.0f;
            for (int j = 0; j < S_; ++j) a2 += sc[row * S_ + j] * vv[j * 3 + d];
            out2[(b * S_ + row) * 3 + d] = sample_coor[(b * S_ + row) * 3 + d] + a2 * inv;
        }
    }
}

// ---------------------------------------------------------------------------
extern "C" void kernel_launch(void* const* d_in, const int* in_sizes, int n_in,
                              void* d_out, int out_size, void* d_ws, size_t ws_size,
                              hipStream_t stream) {
    (void)in_sizes; (void)n_in; (void)out_size; (void)ws_size;
    const float* x    = (const float*)d_in[0];
    const float* coor = (const float*)d_in[1];
    const float* Wq   = (const float*)d_in[2];
    const float* Wk   = (const float*)d_in[3];
    const float* gq   = (const float*)d_in[4];
    const float* bq   = (const float*)d_in[5];
    const float* gk   = (const float*)d_in[6];
    const float* bk   = (const float*)d_in[7];

    float* out1 = (float*)d_out;                 // [B,S,C]
    float* out2 = (float*)d_out + B_ * S_ * C_;  // [B,S,3]

    char* w = (char*)d_ws;
    auto alloc = [&](size_t bytes) -> char* {
        char* p = w;
        w += (bytes + 255) & ~(size_t)255;
        return p;
    };
    int*      fps_idx     = (int*)alloc((size_t)B_ * S_ * 4);
    float*    sample_coor = (float*)alloc((size_t)B_ * S_ * 3 * 4);
    int*      nidx        = (int*)alloc((size_t)B_ * S_ * NS_ * 4);
    float*    diff_x      = (float*)alloc((size_t)B_ * S_ * C_ * 4);
    float*    sample_x    = (float*)alloc((size_t)B_ * S_ * C_ * 4);
    float*    diff_coor   = (float*)alloc((size_t)B_ * S_ * 3 * 4);
    _Float16* Aq          = (_Float16*)alloc((size_t)B_ * S_ * C_ * 2);
    _Float16* Ak          = (_Float16*)alloc((size_t)B_ * S_ * C_ * 2);
    _Float16* Wq16        = (_Float16*)alloc((size_t)C_ * C_ * 2);
    _Float16* Wk16        = (_Float16*)alloc((size_t)C_ * C_ * 2);
    _Float16* q16         = (_Float16*)alloc((size_t)B_ * S_ * C_ * 2);
    _Float16* k16         = (_Float16*)alloc((size_t)B_ * S_ * C_ * 2);

    fps_kernel<<<B_, 1024, 0, stream>>>(coor, fps_idx, sample_coor);
    wconv_kernel<<<(C_ * C_ + 511) / 512, 512, 0, stream>>>(Wq, Wk, Wq16, Wk16);
    ballq_kernel<<<B_ * S_, 32, 0, stream>>>(coor, sample_coor, nidx);
    group_kernel<<<B_ * S_, C_, 0, stream>>>(x, coor, fps_idx, sample_coor, nidx,
                                             out1, diff_x, sample_x, diff_coor);
    ln_kernel<<<2 * B_ * S_, C_, 0, stream>>>(diff_x, sample_x, gq, bq, gk, bk, Aq, Ak);
    qk_gemm_kernel<<<256, 128, 0, stream>>>(Aq, Ak, Wq16, Wk16, q16, k16);
    attn_kernel<<<B_, 512, 0, stream>>>(q16, k16, diff_coor, sample_coor, out2);
}
